// RiboexpModel_13417477833139
// MI455X (gfx1250) — matrix operand order repaired and checked by reading, not compiled, not run on hardware
//
#include <hip/hip_runtime.h>
#include <hip/hip_bf16.h>
#include <math.h>

#define B_N    2048
#define L_SQ   10
#define H_DIM  512
#define HF_DIM 256

typedef __attribute__((ext_vector_type(16))) _Float16 v16h;
typedef __attribute__((ext_vector_type(8)))  _Float16 v8h;
typedef __attribute__((ext_vector_type(8)))  float    v8f;

union UH16 { v16h v; v8h h2[2]; _Float16 h[16]; };
union UF8  { v8f  v; float f[8]; };

// ---------------------------------------------------------------------------
// f16-WMMA GEMM:  C[M,N] = A[M,K] * W[N,K]^T + bias[N]
// A, W are f16 with K padded to a multiple of 32 (zero fill); lda/ldw in halves.
// One wave computes one 16x16 tile. Inner loop: 4x b128 loads + 1 WMMA.
// ---------------------------------------------------------------------------
__global__ void gemm_wmma_f16(const _Float16* __restrict__ A, int lda,
                              const _Float16* __restrict__ W, int ldw,
                              const float* __restrict__ bias,
                              float* __restrict__ C,
                              int M, int N, int K)
{
    const int wave = threadIdx.x >> 5;
    const int lane = threadIdx.x & 31;
    const int nTiles = N >> 4;
    const int totalTiles = (M >> 4) * nTiles;
    const int tile = blockIdx.x * (blockDim.x >> 5) + wave;
    if (tile >= totalTiles) return;

    const int mTile = tile / nTiles;
    const int nTile = tile - mTile * nTiles;

    const int mA  = (mTile << 4) + (lane & 15);     // A row for this lane
    const int kbA = (lane < 16) ? 0 : 16;           // A K sub-base (halves)
    const int nB  = (nTile << 4) + (lane & 15);     // B col (= W row)
    const int kbB = (lane < 16) ? 0 : 16;           // B K sub-base (halves)

    const _Float16* __restrict__ Arow = A + (size_t)mA * (size_t)lda + kbA;
    const _Float16* __restrict__ Wrow = W + (size_t)nB * (size_t)ldw + kbB;

    v8f acc = {0.f, 0.f, 0.f, 0.f, 0.f, 0.f, 0.f, 0.f};

    for (int k0 = 0; k0 < K; k0 += 32) {
        UH16 ua, ub;
        ua.h2[0] = *(const v8h*)(Arow + k0);        // K = kbA + [0..7]
        ua.h2[1] = *(const v8h*)(Arow + k0 + 8);    // K = kbA + [8..15]
        ub.h2[0] = *(const v8h*)(Wrow + k0);        // K = kbB + [0..7]
        ub.h2[1] = *(const v8h*)(Wrow + k0 + 8);    // K = kbB + [8..15]
        acc = __builtin_amdgcn_wmma_f32_16x16x32_f16(
                  false, ua.v, false, ub.v, (short)0, acc, false, false);
    }

    UF8 uc; uc.v = acc;
    const int rowBase = (mTile << 4) + ((lane < 16) ? 0 : 8);
    const int col = (nTile << 4) + (lane & 15);
    const float bv = bias[col];
#pragma unroll
    for (int j = 0; j < 8; ++j)
        C[(size_t)(rowBase + j) * (size_t)N + col] = uc.f[j] + bv;
}

// ---------------------------------------------------------------------------
// Weight conversion: f32 (N x K, row stride ldwSrc) -> f16 (N x Kpad, zero pad)
// ---------------------------------------------------------------------------
__global__ void conv_w_f16_kernel(const float* __restrict__ W, int ldwSrc, int K,
                                  _Float16* __restrict__ O, int Kpad, int total)
{
    int idx = blockIdx.x * blockDim.x + threadIdx.x;
    if (idx >= total) return;
    int n = idx / Kpad, k = idx - n * Kpad;
    O[idx] = (k < K) ? (_Float16)W[(size_t)n * ldwSrc + k] : (_Float16)0.f;
}

// ---------------------------------------------------------------------------
// Elementwise helpers
// ---------------------------------------------------------------------------
__global__ void zero_state_kernel(float* h, _Float16* h16, int n)
{
    int i = blockIdx.x * blockDim.x + threadIdx.x;
    if (i < n) { h[i] = 0.f; h16[i] = (_Float16)0.f; }
}

__global__ void zero_i_kernel(int* p, int n)
{ int i = blockIdx.x * blockDim.x + threadIdx.x; if (i < n) p[i] = 0; }

__device__ __forceinline__ float sigmf(float x) { return 1.f / (1.f + __expf(-x)); }

// GRU gates: gi,gh (2048, 3*Hd) include biases. h (f32) and h16 updated in
// place; hn optionally stored to f16 sequence buffer and/or f32 running max.
__global__ void gru_gate_kernel(const float* __restrict__ gi, const float* __restrict__ gh,
                                float* __restrict__ h, _Float16* __restrict__ h16,
                                _Float16* __restrict__ outSeq, int seqLd,
                                float* __restrict__ outMax, int maxLd, int doMax,
                                int Hd)
{
    int idx = blockIdx.x * blockDim.x + threadIdx.x;
    if (idx >= B_N * Hd) return;
    int m = idx / Hd, j = idx - m * Hd;
    size_t g = (size_t)m * (size_t)(3 * Hd);
    float r  = sigmf(gi[g + j]          + gh[g + j]);
    float z  = sigmf(gi[g + Hd + j]     + gh[g + Hd + j]);
    float nn = tanhf(gi[g + 2 * Hd + j] + r * gh[g + 2 * Hd + j]);
    size_t hi = (size_t)m * Hd + j;
    float hn = (1.f - z) * nn + z * h[hi];
    h[hi]   = hn;
    h16[hi] = (_Float16)hn;
    if (outSeq) outSeq[(size_t)m * seqLd + j] = (_Float16)hn;
    if (outMax) {
        float* o = outMax + (size_t)m * maxLd + j;
        *o = doMax ? fmaxf(*o, hn) : hn;
    }
}

// Policy gate: adds the one-hot(tag) column of gen_ih (f32 original).
__global__ void policy_gate_kernel(const float* __restrict__ aBuf, const float* __restrict__ ghBuf,
                                   float* __restrict__ h, _Float16* __restrict__ h16,
                                   const float* __restrict__ gen_ih,
                                   const int* __restrict__ tag)
{
    int idx = blockIdx.x * blockDim.x + threadIdx.x;
    if (idx >= B_N * H_DIM) return;
    int m = idx >> 9, j = idx & 511;
    int tg = tag[m];                        // 0..9 when read
    size_t g = (size_t)m * 1536;
    const int GIN = 2 * H_DIM + L_SQ + 1;   // 1035
    float ar = aBuf[g + j]        + gen_ih[(size_t)j * GIN          + 1024 + tg];
    float az = aBuf[g + 512 + j]  + gen_ih[(size_t)(512 + j) * GIN  + 1024 + tg];
    float an = aBuf[g + 1024 + j] + gen_ih[(size_t)(1024 + j) * GIN + 1024 + tg];
    float r  = sigmf(ar + ghBuf[g + j]);
    float z  = sigmf(az + ghBuf[g + 512 + j]);
    float nn = tanhf(an + r * ghBuf[g + 1024 + j]);
    size_t hi = (size_t)m * H_DIM + j;
    float hn = (1.f - z) * nn + z * h[hi];
    h[hi]   = hn;
    h16[hi] = (_Float16)hn;
}

// Policy head: 2-class softmax over hn @ cf_w.T + cf_b; one wave per row.
__global__ void policy_head_kernel(const float* __restrict__ h, const float* __restrict__ cfw,
                                   const float* __restrict__ cfb, int* __restrict__ tag,
                                   float* __restrict__ outTags, float* __restrict__ outProbs, int t)
{
    int wave = threadIdx.x >> 5, lane = threadIdx.x & 31;
    int row = blockIdx.x * (blockDim.x >> 5) + wave;
    if (row >= B_N) return;
    const float* hr = h + (size_t)row * H_DIM;
    float s0 = 0.f, s1 = 0.f;
    for (int j = lane; j < H_DIM; j += 32) {
        float v = hr[j];
        s0 += v * cfw[j];
        s1 += v * cfw[H_DIM + j];
    }
#pragma unroll
    for (int o = 16; o > 0; o >>= 1) {
        s0 += __shfl_down(s0, o, 32);
        s1 += __shfl_down(s1, o, 32);
    }
    if (lane == 0) {
        float l0 = s0 + cfb[0], l1 = s1 + cfb[1];
        float mx = fmaxf(l0, l1);
        float e0 = __expf(l0 - mx), e1 = __expf(l1 - mx);
        float p1 = e1 / (e0 + e1);
        int act = (l1 > l0) ? 1 : 0;        // argmax, first-index on ties
        outTags[(size_t)row * L_SQ + t] = (float)act;
        outProbs[(size_t)row * L_SQ + t] = p1;
        tag[row] += act;
    }
}

// Build x16 = [one_hot(input,65) | one_hot(nt,5)x3 | eye(L) | 0-pad] -> (B*L, 96) f16
__global__ void build_x16_kernel(const int* __restrict__ input, const int* __restrict__ nt,
                                 _Float16* __restrict__ x)
{
    int idx = blockIdx.x * blockDim.x + threadIdx.x;
    if (idx >= B_N * L_SQ) return;
    int l = idx % L_SQ;
    _Float16* xr = x + (size_t)idx * 96;
    for (int c = 0; c < 96; ++c) xr[c] = (_Float16)0.f;
    int ch = input[idx];
    if (ch >= 0 && ch < 65) xr[ch] = (_Float16)1.f;
    for (int k = 0; k < 3; ++k) {
        int v = nt[idx * 3 + k];
        if (v >= 0 && v < 5) xr[65 + k * 5 + v] = (_Float16)1.f;
    }
    xr[80 + l] = (_Float16)1.f;
}

// Build xfold16 = [fold(3) | eye(L) | 0-pad] -> (B*L, 32) f16
__global__ void build_xfold16_kernel(const float* __restrict__ fold, _Float16* __restrict__ xf)
{
    int idx = blockIdx.x * blockDim.x + threadIdx.x;
    if (idx >= B_N * L_SQ) return;
    int l = idx % L_SQ;
    _Float16* xr = xf + (size_t)idx * 32;
    for (int c = 0; c < 32; ++c) xr[c] = (_Float16)0.f;
    xr[0] = (_Float16)fold[idx * 3 + 0];
    xr[1] = (_Float16)fold[idx * 3 + 1];
    xr[2] = (_Float16)fold[idx * 3 + 2];
    xr[3 + l] = (_Float16)1.f;
}

// mask_words = tags[:,:,None] * x  (tags are exactly 0/1 -> select)
__global__ void mask_x16_kernel(_Float16* __restrict__ x, const float* __restrict__ tagsOut)
{
    int idx = blockIdx.x * blockDim.x + threadIdx.x;
    if (idx >= B_N * L_SQ * 96) return;
    int bl = idx / 96;
    if (tagsOut[bl] == 0.f) x[idx] = (_Float16)0.f;
}

// Per-row finalize: logit, l2, reward, loss (rlloss == 0 since avg == reward).
__global__ void finalize_kernel(const float* __restrict__ maxbuf, const float* __restrict__ fcw,
                                const float* __restrict__ fcb, const float* __restrict__ target,
                                const int* __restrict__ tag,
                                float* __restrict__ loss, float* __restrict__ reward,
                                float* __restrict__ logitOut)
{
    __shared__ float sd[256];
    int row = blockIdx.x, tid = threadIdx.x;
    const float* r = maxbuf + (size_t)row * 1536;
    float s = 0.f;
    for (int c = tid; c < 1536; c += 256) s += r[c] * fcw[c];
    sd[tid] = s; __syncthreads();
    for (int o = 128; o > 0; o >>= 1) { if (tid < o) sd[tid] += sd[tid + o]; __syncthreads(); }
    if (tid == 0) {
        float logit = sd[0] + fcb[0];
        float d = logit - target[row];
        float l2 = d * d;
        float ts = (float)tag[row];                        // = sum(|tags|)
        reward[row]   = -l2 - 0.1f * fmaxf(ts - 3.0f, 0.f);
        loss[row]     = 0.5f * l2;                         // rlloss = 0 exactly
        logitOut[row] = logit;
    }
}

__global__ void accuracy_kernel(const float* __restrict__ loss, float* __restrict__ acc)
{
    __shared__ float sd[256];
    int tid = threadIdx.x;
    float s = 0.f;
    for (int i = tid; i < B_N; i += 256) s += loss[i] * 2.f;  // l2 = 2*loss
    sd[tid] = s; __syncthreads();
    for (int o = 128; o > 0; o >>= 1) { if (tid < o) sd[tid] += sd[tid + o]; __syncthreads(); }
    if (tid == 0) *acc = sd[0] / (float)B_N;
}

// ---------------------------------------------------------------------------
// Host side
// ---------------------------------------------------------------------------
static inline void launch_gemm(const _Float16* A, int lda, const _Float16* W, int ldw,
                               const float* bias, float* C, int M, int N, int K,
                               hipStream_t s)
{
    int tiles = (M / 16) * (N / 16);
    int blocks = (tiles + 7) / 8;               // 8 waves (tiles) per 256-thread block
    gemm_wmma_f16<<<blocks, 256, 0, s>>>(A, lda, W, ldw, bias, C, M, N, K);
}

static inline void conv_w(const float* W, int N, int K, int ldwSrc,
                          _Float16* dst, int Kpad, hipStream_t s)
{
    int total = N * Kpad;
    conv_w_f16_kernel<<<(total + 255) / 256, 256, 0, s>>>(W, ldwSrc, K, dst, Kpad, total);
}

struct DirW { const _Float16 *ih, *hh; const float *bih, *bhh; };

static void run_bigru(const _Float16* xbase, int Cpad, int Hd,
                      const DirW* w,                 // w[0]=fwd, w[1]=bwd
                      float* hbuf, _Float16* h16, float* gibuf, float* ghbuf,
                      _Float16* seqBase, int seqLd,  // mode A (g_feat), or nullptr
                      float* maxBase, int maxLd,     // mode B (running max), or nullptr
                      int dirStride, hipStream_t s)
{
    const int M = B_N;
    for (int dir = 0; dir < 2; ++dir) {
        zero_state_kernel<<<(M * Hd + 255) / 256, 256, 0, s>>>(hbuf, h16, M * Hd);
        for (int t = 0; t < L_SQ; ++t) {
            int l = dir ? (L_SQ - 1 - t) : t;
            launch_gemm(xbase + (size_t)l * Cpad, L_SQ * Cpad, w[dir].ih, Cpad,
                        w[dir].bih, gibuf, M, 3 * Hd, Cpad, s);
            launch_gemm(h16, Hd, w[dir].hh, Hd, w[dir].bhh, ghbuf, M, 3 * Hd, Hd, s);
            _Float16* oSeq = seqBase ? seqBase + (size_t)l * 1024 + dir * dirStride : nullptr;
            float*    oMax = maxBase ? maxBase + dir * dirStride : nullptr;
            int tot = M * Hd;
            gru_gate_kernel<<<(tot + 255) / 256, 256, 0, s>>>(
                gibuf, ghbuf, hbuf, h16, oSeq, seqLd, oMax, maxLd, (t > 0), Hd);
        }
    }
}

extern "C" void kernel_launch(void* const* d_in, const int* in_sizes, int n_in,
                              void* d_out, int out_size, void* d_ws, size_t ws_size,
                              hipStream_t stream)
{
    (void)in_sizes; (void)n_in; (void)out_size; (void)ws_size;
    const int*   input  = (const int*)  d_in[0];
    const int*   nt     = (const int*)  d_in[1];
    const float* fold   = (const float*)d_in[2];
    const float* target = (const float*)d_in[3];
    const float* P[32];
    for (int i = 0; i < 32; ++i) P[i] = (const float*)d_in[4 + i];
    // params: for pre in (eg,e,f) for d in (f,b): ih,hh,bih,bhh ; then gen/cf/fc
    const float* gen_ih = P[24], *gen_hh = P[25], *gen_bih = P[26], *gen_bhh = P[27];
    const float* cf_w = P[28], *cf_b = P[29], *fc_w = P[30], *fc_b = P[31];

    // ---- output layout (floats) ----
    float* out    = (float*)d_out;
    float* oLoss  = out;                       // 2048
    float* oRew   = out + 2048;                // 2048
    float* oAcc   = out + 4096;                // 1
    float* oLogit = out + 4097;                // 2048
    float* oTags  = out + 6145;                // 2048*10
    float* oProbs = out + 26625;               // 2048*10

    // ---- workspace carve (256B-aligned segments) ----
    char* base = (char*)d_ws;
    auto allocF = [&](size_t n) { float*    p = (float*)base;    base += ((n * 4 + 255) & ~(size_t)255); return p; };
    auto allocH = [&](size_t n) { _Float16* p = (_Float16*)base; base += ((n * 2 + 255) & ~(size_t)255); return p; };

    float* gibuf  = allocF((size_t)B_N * 1536);
    float* ghbuf  = allocF((size_t)B_N * 1536);
    float* hbuf   = allocF((size_t)B_N * H_DIM);
    float* maxbuf = allocF((size_t)B_N * 1536);
    int*   tagbuf = (int*)allocF(B_N);

    _Float16* x16     = allocH((size_t)B_N * L_SQ * 96);   // x / xfold staging
    _Float16* gfeat16 = allocH((size_t)B_N * L_SQ * 1024);
    _Float16* h16     = allocH((size_t)B_N * H_DIM);

    // f16 weight staging (K zero-padded to multiple of 32)
    _Float16* w16[12];                 // eg f/b, e f/b : ih(96), hh ; f f/b : ih(32), hh
    DirW eg[2], e[2], f[2];
    {
        int gi = 0;
        const int cfgC[3]  = { 96, 96, 32 };
        const int cfgK[3]  = { 90, 90, 13 };
        const int cfgH[3]  = { H_DIM, H_DIM, HF_DIM };
        DirW* grp[3] = { eg, e, f };
        for (int g = 0; g < 3; ++g) {
            for (int dir = 0; dir < 2; ++dir) {
                const float* ihW  = P[g * 8 + dir * 4 + 0];
                const float* hhW  = P[g * 8 + dir * 4 + 1];
                const float* bih  = P[g * 8 + dir * 4 + 2];
                const float* bhh  = P[g * 8 + dir * 4 + 3];
                int Hd = cfgH[g];
                _Float16* ih16 = allocH((size_t)3 * Hd * cfgC[g]);
                _Float16* hh16 = allocH((size_t)3 * Hd * Hd);
                conv_w(ihW, 3 * Hd, cfgK[g], cfgK[g], ih16, cfgC[g], stream);
                conv_w(hhW, 3 * Hd, Hd,      Hd,      hh16, Hd,      stream);
                grp[g][dir] = { ih16, hh16, bih, bhh };
                w16[gi++] = ih16; w16[gi++] = hh16;
            }
        }
    }
    _Float16* gen_ih16 = allocH((size_t)1536 * 1024);
    _Float16* gen_hh16 = allocH((size_t)1536 * H_DIM);
    conv_w(gen_ih, 1536, 1024, 2 * H_DIM + L_SQ + 1, gen_ih16, 1024, stream);
    conv_w(gen_hh, 1536, H_DIM, H_DIM,               gen_hh16, H_DIM, stream);

    // 1) build x (f16, padded 96)
    build_x16_kernel<<<(B_N * L_SQ + 255) / 256, 256, 0, stream>>>(input, nt, x16);

    // 2) eg biGRU -> gfeat16 (B, L, 1024) f16
    run_bigru(x16, 96, H_DIM, eg, hbuf, h16, gibuf, ghbuf,
              gfeat16, L_SQ * 1024, nullptr, 0, H_DIM, stream);

    // 3) policy GRU
    zero_state_kernel<<<(B_N * H_DIM + 255) / 256, 256, 0, stream>>>(hbuf, h16, B_N * H_DIM);
    zero_i_kernel<<<(B_N + 255) / 256, 256, 0, stream>>>(tagbuf, B_N);
    for (int t = 0; t < L_SQ; ++t) {
        launch_gemm(gfeat16 + (size_t)t * 1024, L_SQ * 1024,
                    gen_ih16, 1024, gen_bih, gibuf, B_N, 1536, 1024, stream);
        launch_gemm(h16, H_DIM, gen_hh16, H_DIM, gen_bhh, ghbuf, B_N, 1536, H_DIM, stream);
        policy_gate_kernel<<<(B_N * H_DIM + 255) / 256, 256, 0, stream>>>(
            gibuf, ghbuf, hbuf, h16, gen_ih, tagbuf);
        policy_head_kernel<<<B_N / 8, 256, 0, stream>>>(
            hbuf, cf_w, cf_b, tagbuf, oTags, oProbs, t);
    }

    // 4) mask_words = tags * x (in place on x16)
    mask_x16_kernel<<<(B_N * L_SQ * 96 + 255) / 256, 256, 0, stream>>>(x16, oTags);

    // 5) e biGRU -> running max into maxbuf[:, 0:1024]
    run_bigru(x16, 96, H_DIM, e, hbuf, h16, gibuf, ghbuf,
              nullptr, 0, maxbuf, 1536, H_DIM, stream);

    // 6) f biGRU on xfold -> running max into maxbuf[:, 1024:1536]
    build_xfold16_kernel<<<(B_N * L_SQ + 255) / 256, 256, 0, stream>>>(fold, x16);
    run_bigru(x16, 32, HF_DIM, f, hbuf, h16, gibuf, ghbuf,
              nullptr, 0, maxbuf + 1024, 1536, HF_DIM, stream);

    // 7) finalize + deterministic accuracy reduction
    finalize_kernel<<<B_N, 256, 0, stream>>>(maxbuf, fc_w, fc_b, target, tagbuf,
                                             oLoss, oRew, oLogit);
    accuracy_kernel<<<1, 256, 0, stream>>>(oLoss, oAcc);
}
